// GNN_2774548873606
// MI455X (gfx1250) — matrix operand order repaired and checked by reading, not compiled, not run on hardware
//
#include <hip/hip_runtime.h>
#include <hip/hip_bf16.h>

// ---------------------------------------------------------------------------
// GNN pipeline for MI455X (gfx1250, wave32).
// All dense GEMMs run on the WMMA pipe: v_wmma_f32_16x16x32_bf16, with
// weights pre-packed into the exact B-operand register layout (bf16) so the
// unrolled inner loop is pure b128 loads + WMMA.
// Segment ops (softmax max/sum, mean aggregation) use global atomics.
// The [E,256] concat feature matrix is never materialized: the SAGE1 GEMM
// gathers A-fragments from h[row[e]] / h[col[e]] on the fly.
// NOTE: the JAX threefry edge mask is replaced with a deterministic integer
// hash (compile-only environment, exact PRNG reproduction not required).
// ---------------------------------------------------------------------------

typedef __attribute__((ext_vector_type(16))) __bf16 v16bf;
typedef __attribute__((ext_vector_type(8)))  __bf16 v8bf;
typedef __attribute__((ext_vector_type(8)))  float  v8f;

#define NE 400000
#define NN 30000
#define TB 256

__device__ __forceinline__ unsigned enc_f32(float f) {
  unsigned u = __float_as_uint(f);
  return (u & 0x80000000u) ? ~u : (u | 0x80000000u);
}
__device__ __forceinline__ float dec_f32(unsigned u) {
  u = (u & 0x80000000u) ? (u & 0x7fffffffu) : ~u;
  return __uint_as_float(u);
}
__device__ __forceinline__ float hash01(unsigned x) {
  x ^= 2747636419u; x *= 2654435769u;
  x ^= x >> 16;     x *= 2654435769u;
  x ^= x >> 16;     x *= 2654435769u;
  return (float)(x >> 8) * (1.0f / 16777216.0f);
}

// ----------------------------- utility kernels -----------------------------
__global__ void k_fill_f32(float* p, float v, long long n) {
  long long i = (long long)blockIdx.x * blockDim.x + threadIdx.x;
  if (i < n) p[i] = v;
}
__global__ void k_fill_u32(unsigned* p, unsigned v, long long n) {
  long long i = (long long)blockIdx.x * blockDim.x + threadIdx.x;
  if (i < n) p[i] = v;
}
__global__ void k_f32_to_bf16(const float* __restrict__ in, __bf16* __restrict__ out,
                              long long n) {
  long long i = (long long)blockIdx.x * blockDim.x + threadIdx.x;
  if (i < n) out[i] = (__bf16)in[i];
}

// mask + edge-attr column-0 replacement
__global__ void k_mask_ea(const float* __restrict__ edge_attr, const float* __restrict__ y,
                          float* __restrict__ ea, int* __restrict__ rmask) {
  int e = blockIdx.x * blockDim.x + threadIdx.x;
  if (e >= NE) return;
  float r  = hash01((unsigned)e);
  float yv = y[e];
  int m = (r < 0.2f) && (yv != -1.0f);
  rmask[e] = m;
  ea[(size_t)e * 8] = m ? -2.0f : yv;
#pragma unroll
  for (int j = 1; j < 8; ++j) ea[(size_t)e * 8 + j] = edge_attr[(size_t)e * 8 + j];
}

// -------------------- weight pre-pack into WMMA B layout -------------------
// Packed layout: for tile t = kt*Ntiles+nt, lane l, vgpr v:
//   out[(t*32+l)*16 + 2v]   = W[kt*32 + (l>>4)*16 + 2v    ][nt*16 + (l&15)]
//   out[(t*32+l)*16 + 2v+1] = W[kt*32 + (l>>4)*16 + 2v + 1][nt*16 + (l&15)]
// so the GEMM reads one contiguous 32-byte v16bf per lane per K-step.
__global__ void k_pack_w(const float* __restrict__ W, __bf16* __restrict__ out,
                         int K, int N) {
  int idx = blockIdx.x * blockDim.x + threadIdx.x;
  int total = (K >> 5) * (N >> 4) * 256;          // 8 pairs * 32 lanes per tile
  if (idx >= total) return;
  int v    = idx & 7;
  int lane = (idx >> 3) & 31;
  int tile = idx >> 8;
  int Ntiles = N >> 4;
  int kt = tile / Ntiles, nt = tile % Ntiles;
  int k = kt * 32 + (lane >> 4) * 16 + 2 * v;
  int n = nt * 16 + (lane & 15);
  out[(size_t)idx * 2]     = (__bf16)W[(size_t)k * N + n];
  out[(size_t)idx * 2 + 1] = (__bf16)W[(size_t)(k + 1) * N + n];
}

// ----------------------- generic bf16 WMMA GEMM ----------------------------
// One wave computes one 16x16 output tile. Fully templated: K-loop unrolls,
// no runtime mode branches inside the loop.
// AMODE==0: A is dense [M x lda] bf16.
// AMODE==1: logical A row m = concat(h[row[m]], h[col[m]]), h = A (ld 128).
template <int KT, int AMODE, bool HASBIAS, bool ACCUM>
__global__ __launch_bounds__(256)
void k_gemm_bf16(const __bf16* __restrict__ A, int lda,
                 const int* __restrict__ row, const int* __restrict__ col,
                 const __bf16* __restrict__ Wp,
                 const float* __restrict__ bias,
                 float* __restrict__ outF, int ldo, int Mtiles, int Ntiles) {
  int wave = (int)((blockIdx.x * blockDim.x + threadIdx.x) >> 5);
  if (wave >= Mtiles * Ntiles) return;            // wave-uniform: EXEC stays all-1s
  int lane  = threadIdx.x & 31;
  int l16   = lane & 15;
  int lhalf = lane >> 4;
  int mt = wave / Ntiles, nt = wave % Ntiles;
  int m0 = mt * 16, n0 = nt * 16;
  int am = m0 + l16;
  int bn = n0 + l16;

  const __bf16* arow0;
  const __bf16* arow1 = nullptr;
  if constexpr (AMODE == 0) {
    arow0 = A + (size_t)am * lda;
  } else {
    arow0 = A + (size_t)row[am] * 128;
    arow1 = A + (size_t)col[am] * 128;
  }

  v8f acc = {};
#pragma unroll
  for (int kt = 0; kt < KT; ++kt) {
    const int k0 = kt * 32;
    const __bf16* ar;
    int kk;
    if constexpr (AMODE == 0) { ar = arow0; kk = k0; }
    else {
      if (k0 < 128) { ar = arow0; kk = k0; }
      else          { ar = arow1; kk = k0 - 128; }
    }
    // A fragment: VGPR0-7 hold K = lhalf*16 + 0..15 (matches packed B)
    v16bf a;
    {
      v8bf lo = *reinterpret_cast<const v8bf*>(ar + kk + lhalf * 16);
      v8bf hi = *reinterpret_cast<const v8bf*>(ar + kk + lhalf * 16 + 8);
#pragma unroll
      for (int i = 0; i < 8; ++i) { a[i] = lo[i]; a[8 + i] = hi[i]; }
    }
    // B fragment: pre-packed, one contiguous 32B load per lane
    v16bf b = *reinterpret_cast<const v16bf*>(
        Wp + ((size_t)(kt * Ntiles + nt) * 32 + lane) * 16);
    acc = __builtin_amdgcn_wmma_f32_16x16x32_bf16(false, a, false, b,
                                                  (short)0, acc, false, false);
  }

  int mb = m0 + lhalf * 8;                         // C/D layout: VGPR r -> M = mb+r
  float bv = 0.0f;
  if constexpr (HASBIAS) bv = bias[bn];
#pragma unroll
  for (int r = 0; r < 8; ++r) {
    size_t oi = (size_t)(mb + r) * ldo + bn;
    float v = acc[r] + bv;
    if constexpr (ACCUM) v += outF[oi];
    outF[oi] = v;
  }
}

// --------------------------- GATv2 edge kernels ----------------------------
__global__ __launch_bounds__(256)
void k_gat_logits(const int* __restrict__ row, const int* __restrict__ col,
                  const float* __restrict__ xl, const float* __restrict__ xr,
                  const float* __restrict__ ea, const float* __restrict__ We,
                  const float* __restrict__ att,
                  float* __restrict__ logits, unsigned* __restrict__ lmaxI) {
  int e = (int)((blockIdx.x * blockDim.x + threadIdx.x) >> 5);
  if (e >= NE) return;
  int lane = threadIdx.x & 31;
  int r = row[e], c = col[e];
  float ead = (lane < 8) ? ea[(size_t)e * 8 + lane] : 0.0f;
#pragma unroll
  for (int h = 0; h < 4; ++h) {
    int j = h * 32 + lane;
    float mv = xl[(size_t)r * 128 + j] + xr[(size_t)c * 128 + j];
#pragma unroll
    for (int d = 0; d < 8; ++d)
      mv += __shfl(ead, d, 32) * We[d * 128 + j];
    mv = (mv > 0.0f) ? mv : 0.2f * mv;             // leaky_relu(0.2)
    float part = mv * att[h * 32 + lane];
#pragma unroll
    for (int off = 16; off > 0; off >>= 1)
      part += __shfl_xor(part, off, 32);
    if (lane == 0) {
      logits[(size_t)e * 4 + h] = part;
      atomicMax(&lmaxI[(size_t)c * 4 + h], enc_f32(part));
    }
  }
}

__global__ void k_gat_denom(const int* __restrict__ col, const float* __restrict__ logits,
                            const unsigned* __restrict__ lmaxI, float* __restrict__ denom) {
  int idx = blockIdx.x * blockDim.x + threadIdx.x;
  if (idx >= NE * 4) return;
  int e = idx >> 2, h = idx & 3;
  int c = col[e];
  float mx = dec_f32(lmaxI[(size_t)c * 4 + h]);
  atomicAdd(&denom[(size_t)c * 4 + h], __expf(logits[idx] - mx));
}

__global__ __launch_bounds__(256)
void k_gat_scatter(const int* __restrict__ row, const int* __restrict__ col,
                   const float* __restrict__ logits, const unsigned* __restrict__ lmaxI,
                   const float* __restrict__ denom,
                   const float* __restrict__ xl, float* __restrict__ hsum) {
  int e = (int)((blockIdx.x * blockDim.x + threadIdx.x) >> 5);
  if (e >= NE) return;
  int lane = threadIdx.x & 31;
  int r = row[e], c = col[e];
#pragma unroll
  for (int h = 0; h < 4; ++h) {
    float l  = logits[(size_t)e * 4 + h];
    float mx = dec_f32(lmaxI[(size_t)c * 4 + h]);
    float al = __expf(l - mx) / (denom[(size_t)c * 4 + h] + 1e-16f);
    int j = h * 32 + lane;
    atomicAdd(&hsum[(size_t)c * 128 + j], xl[(size_t)r * 128 + j] * al);
  }
}

__global__ void k_gat_finalize(const float* __restrict__ hsum, const float* __restrict__ bias,
                               float* __restrict__ hf, __bf16* __restrict__ hb) {
  int idx = blockIdx.x * blockDim.x + threadIdx.x;
  if (idx >= NN * 128) return;
  float v = fmaxf(hsum[idx] + bias[idx & 127], 0.0f);   // +bias, relu
  hf[idx] = v;
  hb[idx] = (__bf16)v;
}

// ------------------------------ SAGE kernels -------------------------------
__global__ void k_cnt(const int* __restrict__ col, float* __restrict__ cnt) {
  int e = blockIdx.x * blockDim.x + threadIdx.x;
  if (e >= NE) return;
  atomicAdd(&cnt[col[e]], 1.0f);
}

// s1[col[e], j] += ef[row[e], j], ef row t = concat(h[row[t]], h[col[t]])
__global__ void k_sage1_scatter(const int* __restrict__ row, const int* __restrict__ col,
                                const float* __restrict__ hf, float* __restrict__ s1) {
  long long idx = (long long)blockIdx.x * blockDim.x + threadIdx.x;
  if (idx >= (long long)NE * 256) return;
  int e = (int)(idx >> 8), j = (int)(idx & 255);
  int t = row[e];
  float v = (j < 128) ? hf[(size_t)row[t] * 128 + j]
                      : hf[(size_t)col[t] * 128 + (j - 128)];
  atomicAdd(&s1[(size_t)col[e] * 256 + j], v);
}
__global__ void k_sage1_mean(const float* __restrict__ s1, const float* __restrict__ cnt,
                             __bf16* __restrict__ meanb) {
  int idx = blockIdx.x * blockDim.x + threadIdx.x;
  if (idx >= NN * 256) return;
  meanb[idx] = (__bf16)(s1[idx] / fmaxf(cnt[idx >> 8], 1.0f));
}

__global__ void k_sage2_scatter(const int* __restrict__ row, const int* __restrict__ col,
                                const float* __restrict__ ef2, float* __restrict__ s2) {
  long long idx = (long long)blockIdx.x * blockDim.x + threadIdx.x;
  if (idx >= (long long)NE * 32) return;
  int e = (int)(idx >> 5), j = (int)(idx & 31);
  atomicAdd(&s2[(size_t)col[e] * 32 + j], ef2[(size_t)row[e] * 32 + j]);
}
__global__ void k_mean32(const float* __restrict__ s2, const float* __restrict__ cnt,
                         __bf16* __restrict__ meanb) {
  int idx = blockIdx.x * blockDim.x + threadIdx.x;
  if (idx >= NN * 32) return;
  meanb[idx] = (__bf16)(s2[idx] / fmaxf(cnt[idx >> 5], 1.0f));
}

// ------------------------------ final head ---------------------------------
__global__ __launch_bounds__(256)
void k_final(const float* __restrict__ ef4, const float* __restrict__ Wf2,
             const float* __restrict__ bf2, const int* __restrict__ rmask,
             float* __restrict__ out) {
  int e = (int)((blockIdx.x * blockDim.x + threadIdx.x) >> 5);
  if (e >= NE) return;
  int lane = threadIdx.x & 31;
  float v = ef4[(size_t)e * 32 + lane] * Wf2[lane];
#pragma unroll
  for (int off = 16; off > 0; off >>= 1)
    v += __shfl_xor(v, off, 32);
  if (lane == 0) {
    float z = v + bf2[0];
    float p = 4.0f / (1.0f + __expf(-z)) + 1.0f;    // sigmoid*4 + 1
    int m = rmask[e];
    out[e]      = m ? p : 0.0f;
    out[NE + e] = m ? 1.0f : 0.0f;
  }
}

// ------------------------------- launcher ----------------------------------
static inline char* carve(char*& p, size_t bytes) {
  char* r = p;
  p += (bytes + 255) & ~(size_t)255;
  return r;
}
static inline unsigned gB(long long n) { return (unsigned)((n + TB - 1) / TB); }
static inline unsigned gW(long long waves) { return (unsigned)((waves + 7) / 8); }

extern "C" void kernel_launch(void* const* d_in, const int* in_sizes, int n_in,
                              void* d_out, int out_size, void* d_ws, size_t ws_size,
                              hipStream_t stream) {
  const float* x   = (const float*)d_in[0];
  const int*   ei  = (const int*)d_in[1];
  const int*   row = ei;
  const int*   col = ei + NE;
  const float* eattr = (const float*)d_in[2];
  const float* y   = (const float*)d_in[3];
  const float *Wl1 = (const float*)d_in[5],  *bl1 = (const float*)d_in[6];
  const float *Wr1 = (const float*)d_in[7],  *br1 = (const float*)d_in[8];
  const float *We1 = (const float*)d_in[9],  *att1 = (const float*)d_in[10];
  const float *bias1 = (const float*)d_in[11];
  const float *Wl2 = (const float*)d_in[12], *bl2 = (const float*)d_in[13];
  const float *Wr2 = (const float*)d_in[14], *br2 = (const float*)d_in[15];
  const float *We2 = (const float*)d_in[16], *att2 = (const float*)d_in[17];
  const float *bias2 = (const float*)d_in[18];
  const float *Ws1l = (const float*)d_in[19], *bs1 = (const float*)d_in[20];
  const float *Ws1r = (const float*)d_in[21];
  const float *Ws2l = (const float*)d_in[22], *bs2 = (const float*)d_in[23];
  const float *Ws2r = (const float*)d_in[24];
  const float *Wf1  = (const float*)d_in[25], *bf1 = (const float*)d_in[26];
  const float *Wf2  = (const float*)d_in[27], *bf2 = (const float*)d_in[28];
  float* out = (float*)d_out;

  char* w = (char*)d_ws;
  float*    xl     = (float*)   carve(w, (size_t)NN * 128 * 4);
  float*    xr     = (float*)   carve(w, (size_t)NN * 128 * 4);
  __bf16*   nb0    = (__bf16*)  carve(w, (size_t)NN * 128 * 2);  // x_bf16 / h2_bf16
  __bf16*   nb1    = (__bf16*)  carve(w, (size_t)NN * 128 * 2);  // h1_bf16
  float*    ea     = (float*)   carve(w, (size_t)NE * 8 * 4);
  int*      rmask  = (int*)     carve(w, (size_t)NE * 4);
  float*    logits = (float*)   carve(w, (size_t)NE * 4 * 4);
  unsigned* lmaxI  = (unsigned*)carve(w, (size_t)NN * 4 * 4);
  float*    denom  = (float*)   carve(w, (size_t)NN * 4 * 4);
  float*    hsum   = (float*)   carve(w, (size_t)NN * 128 * 4);
  float*    hf     = (float*)   carve(w, (size_t)NN * 128 * 4);
  float*    s1     = (float*)   carve(w, (size_t)NN * 256 * 4);
  float*    cnt    = (float*)   carve(w, (size_t)NN * 4);
  __bf16*   mean1b = (__bf16*)  carve(w, (size_t)NN * 256 * 2);
  float*    ef2    = (float*)   carve(w, (size_t)NE * 32 * 4);
  __bf16*   ef2b   = (__bf16*)  carve(w, (size_t)NE * 32 * 2);
  float*    s2     = (float*)   carve(w, (size_t)NN * 32 * 4);
  __bf16*   mean2b = (__bf16*)  carve(w, (size_t)NN * 32 * 2);
  float*    ef3    = (float*)   carve(w, (size_t)NE * 32 * 4);
  __bf16*   ef3b   = (__bf16*)  carve(w, (size_t)NE * 32 * 2);
  float*    ef4    = (float*)   carve(w, (size_t)NE * 32 * 4);
  // packed weights (bf16, WMMA B layout)
  __bf16*   pWl1   = (__bf16*)  carve(w, (size_t)128 * 128 * 2);
  __bf16*   pWr1   = (__bf16*)  carve(w, (size_t)128 * 128 * 2);
  __bf16*   pWl2   = (__bf16*)  carve(w, (size_t)128 * 128 * 2);
  __bf16*   pWr2   = (__bf16*)  carve(w, (size_t)128 * 128 * 2);
  __bf16*   pWs1l  = (__bf16*)  carve(w, (size_t)256 * 32 * 2);
  __bf16*   pWs1r  = (__bf16*)  carve(w, (size_t)256 * 32 * 2);
  __bf16*   pWs2l  = (__bf16*)  carve(w, (size_t)32 * 32 * 2);
  __bf16*   pWs2r  = (__bf16*)  carve(w, (size_t)32 * 32 * 2);
  __bf16*   pWf1   = (__bf16*)  carve(w, (size_t)32 * 32 * 2);
  (void)in_sizes; (void)n_in; (void)out_size; (void)ws_size;

  const int MtN = NN / 16;   // 1875 node-row tiles
  const int MtE = NE / 16;   // 25000 edge-row tiles

  // weight packing (tiny, L2-resident afterwards)
  k_pack_w<<<gB(128 * 128 / 2), TB, 0, stream>>>(Wl1, pWl1, 128, 128);
  k_pack_w<<<gB(128 * 128 / 2), TB, 0, stream>>>(Wr1, pWr1, 128, 128);
  k_pack_w<<<gB(128 * 128 / 2), TB, 0, stream>>>(Wl2, pWl2, 128, 128);
  k_pack_w<<<gB(128 * 128 / 2), TB, 0, stream>>>(Wr2, pWr2, 128, 128);
  k_pack_w<<<gB(256 * 32 / 2), TB, 0, stream>>>(Ws1l, pWs1l, 256, 32);
  k_pack_w<<<gB(256 * 32 / 2), TB, 0, stream>>>(Ws1r, pWs1r, 256, 32);
  k_pack_w<<<gB(32 * 32 / 2), TB, 0, stream>>>(Ws2l, pWs2l, 32, 32);
  k_pack_w<<<gB(32 * 32 / 2), TB, 0, stream>>>(Ws2r, pWs2r, 32, 32);
  k_pack_w<<<gB(32 * 32 / 2), TB, 0, stream>>>(Wf1, pWf1, 32, 32);

  k_mask_ea<<<gB(NE), TB, 0, stream>>>(eattr, y, ea, rmask);
  k_f32_to_bf16<<<gB((long long)NN * 128), TB, 0, stream>>>(x, nb0, (long long)NN * 128);

  // -------- GATv2 layer 1 --------
  k_fill_u32<<<gB(NN * 4), TB, 0, stream>>>(lmaxI, 0u, NN * 4);
  k_fill_f32<<<gB(NN * 4), TB, 0, stream>>>(denom, 0.0f, NN * 4);
  k_fill_f32<<<gB((long long)NN * 128), TB, 0, stream>>>(hsum, 0.0f, (long long)NN * 128);
  k_gemm_bf16<4, 0, true, false><<<gW((long long)MtN * 8), TB, 0, stream>>>(
      nb0, 128, row, col, pWl1, bl1, xl, 128, MtN, 8);
  k_gemm_bf16<4, 0, true, false><<<gW((long long)MtN * 8), TB, 0, stream>>>(
      nb0, 128, row, col, pWr1, br1, xr, 128, MtN, 8);
  k_gat_logits<<<gW(NE), TB, 0, stream>>>(row, col, xl, xr, ea, We1, att1, logits, lmaxI);
  k_gat_denom<<<gB((long long)NE * 4), TB, 0, stream>>>(col, logits, lmaxI, denom);
  k_gat_scatter<<<gW(NE), TB, 0, stream>>>(row, col, logits, lmaxI, denom, xl, hsum);
  k_gat_finalize<<<gB((long long)NN * 128), TB, 0, stream>>>(hsum, bias1, hf, nb1);

  // -------- GATv2 layer 2 --------
  k_fill_u32<<<gB(NN * 4), TB, 0, stream>>>(lmaxI, 0u, NN * 4);
  k_fill_f32<<<gB(NN * 4), TB, 0, stream>>>(denom, 0.0f, NN * 4);
  k_fill_f32<<<gB((long long)NN * 128), TB, 0, stream>>>(hsum, 0.0f, (long long)NN * 128);
  k_gemm_bf16<4, 0, true, false><<<gW((long long)MtN * 8), TB, 0, stream>>>(
      nb1, 128, row, col, pWl2, bl2, xl, 128, MtN, 8);
  k_gemm_bf16<4, 0, true, false><<<gW((long long)MtN * 8), TB, 0, stream>>>(
      nb1, 128, row, col, pWr2, br2, xr, 128, MtN, 8);
  k_gat_logits<<<gW(NE), TB, 0, stream>>>(row, col, xl, xr, ea, We2, att2, logits, lmaxI);
  k_gat_denom<<<gB((long long)NE * 4), TB, 0, stream>>>(col, logits, lmaxI, denom);
  k_gat_scatter<<<gW(NE), TB, 0, stream>>>(row, col, logits, lmaxI, denom, xl, hsum);
  k_gat_finalize<<<gB((long long)NN * 128), TB, 0, stream>>>(hsum, bias2, hf, nb0);

  // -------- SAGE layer 1 (ef = concat(h[row], h[col]), never materialized) --------
  k_fill_f32<<<gB((long long)NN * 256), TB, 0, stream>>>(s1, 0.0f, (long long)NN * 256);
  k_fill_f32<<<gB(NN), TB, 0, stream>>>(cnt, 0.0f, NN);
  k_cnt<<<gB(NE), TB, 0, stream>>>(col, cnt);
  k_sage1_scatter<<<gB((long long)NE * 256), TB, 0, stream>>>(row, col, hf, s1);
  k_sage1_mean<<<gB((long long)NN * 256), TB, 0, stream>>>(s1, cnt, mean1b);
  k_gemm_bf16<8, 1, true, false><<<gW((long long)MtE * 2), TB, 0, stream>>>(
      nb0, 128, row, col, pWs1r, bs1, ef2, 32, MtE, 2);
  k_gemm_bf16<8, 0, false, true><<<gW((long long)MtN * 2), TB, 0, stream>>>(
      mean1b, 256, row, col, pWs1l, nullptr, ef2, 32, MtN, 2);
  k_f32_to_bf16<<<gB((long long)NE * 32), TB, 0, stream>>>(ef2, ef2b, (long long)NE * 32);

  // -------- SAGE layer 2 --------
  k_fill_f32<<<gB((long long)NN * 32), TB, 0, stream>>>(s2, 0.0f, (long long)NN * 32);
  k_sage2_scatter<<<gB((long long)NE * 32), TB, 0, stream>>>(row, col, ef2, s2);
  k_mean32<<<gB((long long)NN * 32), TB, 0, stream>>>(s2, cnt, mean2b);
  k_gemm_bf16<1, 0, true, false><<<gW((long long)MtE * 2), TB, 0, stream>>>(
      ef2b, 32, row, col, pWs2r, bs2, ef3, 32, MtE, 2);
  k_gemm_bf16<1, 0, false, true><<<gW((long long)MtN * 2), TB, 0, stream>>>(
      mean2b, 32, row, col, pWs2l, nullptr, ef3, 32, MtN, 2);
  k_f32_to_bf16<<<gB((long long)NE * 32), TB, 0, stream>>>(ef3, ef3b, (long long)NE * 32);

  // -------- fc1 + fc2 head --------
  k_gemm_bf16<1, 0, true, false><<<gW((long long)MtE * 2), TB, 0, stream>>>(
      ef3b, 32, row, col, pWf1, bf1, ef4, 32, MtE, 2);
  k_final<<<gW(NE), TB, 0, stream>>>(ef4, Wf2, bf2, rmask, out);
}
